// SlotAttention_37821482008789
// MI455X (gfx1250) — matrix operand order repaired and checked by reading, not compiled, not run on hardware
//
#include <hip/hip_runtime.h>
#include <hip/hip_bf16.h>
#include <math.h>

#define B_ 32
#define N_ 4096
#define K_ 8
#define C_ 256
#define F_ 1024

typedef __bf16 bf16;
typedef __attribute__((ext_vector_type(8)))  bf16  v8bf;
typedef __attribute__((ext_vector_type(16))) bf16  v16bf;
typedef __attribute__((ext_vector_type(8)))  float v8f;

__device__ __forceinline__ v16bf cat8(v8bf lo, v8bf hi) {
  return __builtin_shufflevector(lo, hi, 0,1,2,3,4,5,6,7,8,9,10,11,12,13,14,15);
}
__device__ __forceinline__ v16bf load16bf(const bf16* p) {
  return cat8(*(const v8bf*)p, *(const v8bf*)(p + 8));
}
__device__ __forceinline__ v8f wmma_bf16(v16bf a, v16bf b, v8f c) {
  return __builtin_amdgcn_wmma_f32_16x16x32_bf16(false, a, false, b, (short)0, c, false, false);
}
__device__ __forceinline__ v8f vzero8() {
  v8f z; for (int j = 0; j < 8; ++j) z[j] = 0.f; return z;
}
// A-fragment (16x32 bf16) from row-major bf16 row pointer already offset by kt*32
__device__ __forceinline__ v16bf load_a_bf16(const bf16* rowk, int hi) {
  return cat8(*(const v8bf*)(rowk + hi*16), *(const v8bf*)(rowk + hi*16 + 8));
}
// A-fragment from row-major f32 (converted on the fly, optional per-row scale)
__device__ __forceinline__ v16bf load_a_f32(const float* rowk, int hi, float scale) {
  const float* p  = rowk + hi*16;
  float4 q0 = *(const float4*)p,        q1 = *(const float4*)(p + 4);
  float4 q2 = *(const float4*)(p + 8),  q3 = *(const float4*)(p + 12);
  float vv[16] = {q0.x,q0.y,q0.z,q0.w, q1.x,q1.y,q1.z,q1.w,
                  q2.x,q2.y,q2.z,q2.w, q3.x,q3.y,q3.z,q3.w};
  v16bf a;
  for (int i = 0; i < 16; ++i) a[i] = (bf16)(vv[i] * scale);
  return a;
}
__device__ __forceinline__ void wave_red2(float& s, float& s2) {
  for (int m = 16; m >= 1; m >>= 1) {
    s  += __shfl_xor(s,  m, 32);
    s2 += __shfl_xor(s2, m, 32);
  }
}

// ---------------- weight f32 -> bf16 ----------------
__global__ void cvt_kernel(const float* __restrict__ s, bf16* __restrict__ d, int n) {
  int i = blockIdx.x * blockDim.x + threadIdx.x;
  if (i < n) d[i] = (bf16)s[i];
}

// ---------------- fused LN(x) + k/v projection ----------------
// grid: (B*N/64) blocks x 128 threads (4 waves, 16 rows each)
__global__ void __launch_bounds__(128) kv_kernel(
    const float* __restrict__ x, const float* __restrict__ g, const float* __restrict__ bb,
    const bf16* __restrict__ Wk, const bf16* __restrict__ Wv,
    bf16* __restrict__ kout, bf16* __restrict__ vT) {
  __shared__ __align__(16) bf16 lds[4][16 * 272];
  const int wave = threadIdx.x >> 5;
  const int l = threadIdx.x & 31, m16 = l & 15, hi = l >> 4;
  const int rowbase = blockIdx.x * 64 + wave * 16;

  for (int r = 0; r < 16; ++r) {
    const float* xr = x + (size_t)(rowbase + r) * C_ + l * 8;
    float4 p0 = *(const float4*)xr, p1 = *(const float4*)(xr + 4);
    float v[8] = {p0.x,p0.y,p0.z,p0.w,p1.x,p1.y,p1.z,p1.w};
    float s = 0.f, s2 = 0.f;
    for (int i = 0; i < 8; ++i) { s += v[i]; s2 += v[i]*v[i]; }
    wave_red2(s, s2);
    float mean = s * (1.f / C_);
    float rstd = rsqrtf(s2 * (1.f / C_) - mean * mean + 1e-5f);
    for (int i = 0; i < 8; ++i) {
      int c = l * 8 + i;
      lds[wave][r * 272 + c] = (bf16)((v[i] - mean) * rstd * g[c] + bb[c]);
    }
  }
  __syncthreads();

  v16bf a[8];
  for (int kt = 0; kt < 8; ++kt)
    a[kt] = load_a_bf16(&lds[wave][m16 * 272 + kt * 32], hi);

  const int b = rowbase >> 12, n0 = rowbase & (N_ - 1);
  for (int mat = 0; mat < 2; ++mat) {
    const bf16* W = mat ? Wv : Wk;
    for (int nt = 0; nt < 16; ++nt) {
      v8f acc = vzero8();
      const bf16* wrow = W + (size_t)(nt * 16 + m16) * C_;
      for (int kt = 0; kt < 8; ++kt)
        acc = wmma_bf16(a[kt], load16bf(wrow + kt * 32 + hi * 16), acc);
      if (mat == 0) {   // k natural [b][n][c]
        for (int j = 0; j < 8; ++j)
          kout[(size_t)(rowbase + hi * 8 + j) * C_ + nt * 16 + m16] = (bf16)acc[j];
      } else {          // v transposed [b][c][n] -> contiguous v8bf store
        v8bf pk; for (int j = 0; j < 8; ++j) pk[j] = (bf16)acc[j];
        *(v8bf*)(vT + (size_t)b * C_ * N_ + (size_t)(nt * 16 + m16) * N_ + n0 + hi * 8) = pk;
      }
    }
  }
}

// ---------------- LN(slots) + q projection (scale folded) ----------------
// grid: dim3(16 /*ntile*/, B) x 32
__global__ void __launch_bounds__(32) q_kernel(
    const float* __restrict__ slots, const float* __restrict__ g, const float* __restrict__ bb,
    const bf16* __restrict__ Wq, bf16* __restrict__ qbf, float* __restrict__ colsum) {
  __shared__ __align__(16) bf16 lds[16 * 272];
  const int l = threadIdx.x & 31, m16 = l & 15, hi = l >> 4;
  const int nt = blockIdx.x, b = blockIdx.y;
  if (nt == 0 && l < K_) colsum[b * K_ + l] = 0.f;

  for (int s = 0; s < 16; ++s) {
    if (s < K_) {
      const float* xr = slots + (size_t)(b * K_ + s) * C_ + l * 8;
      float4 p0 = *(const float4*)xr, p1 = *(const float4*)(xr + 4);
      float v[8] = {p0.x,p0.y,p0.z,p0.w,p1.x,p1.y,p1.z,p1.w};
      float su = 0.f, s2 = 0.f;
      for (int i = 0; i < 8; ++i) { su += v[i]; s2 += v[i]*v[i]; }
      wave_red2(su, s2);
      float mean = su * (1.f / C_);
      float rstd = rsqrtf(s2 * (1.f / C_) - mean * mean + 1e-5f);
      for (int i = 0; i < 8; ++i) {
        int c = l * 8 + i;
        lds[s * 272 + c] = (bf16)((v[i] - mean) * rstd * g[c] + bb[c]);
      }
    } else {
      for (int i = 0; i < 8; ++i) lds[s * 272 + l * 8 + i] = (bf16)0.f;
    }
  }
  __syncthreads();

  v8f acc = vzero8();
  const bf16* wrow = Wq + (size_t)(nt * 16 + m16) * C_;
  for (int kt = 0; kt < 8; ++kt)
    acc = wmma_bf16(load_a_bf16(&lds[m16 * 272 + kt * 32], hi),
                    load16bf(wrow + kt * 32 + hi * 16), acc);
  for (int j = 0; j < 8; ++j)   // q * C^-0.5, padded rows 8..15 are zeros
    qbf[(size_t)b * 16 * C_ + (size_t)(hi * 8 + j) * C_ + nt * 16 + m16] =
        (bf16)(acc[j] * 0.0625f);
}

// ---------------- logits + softmax over slots + attn out + colsum ----------------
// grid: dim3(N/16, B) x 32
__global__ void __launch_bounds__(32) logits_kernel(
    const bf16* __restrict__ kbf, const bf16* __restrict__ qbf,
    float* __restrict__ attn, float* __restrict__ colsum) {
  const int l = threadIdx.x & 31, m16 = l & 15, hi = l >> 4;
  const int nt = blockIdx.x, b = blockIdx.y, n0 = nt * 16;
  v8f acc = vzero8();
  const bf16* krow = kbf + ((size_t)b * N_ + n0 + m16) * C_;
  const bf16* qrow = qbf + (size_t)b * 16 * C_ + (size_t)m16 * C_;
  for (int kt = 0; kt < 8; ++kt)
    acc = wmma_bf16(load_a_bf16(krow + kt * 32, hi),
                    load16bf(qrow + kt * 32 + hi * 16), acc);
  // softmax across slot columns: lanes {s, s^1, s^2, s^4} within groups of 8
  v8f mx = acc;
  for (int d = 1; d < 8; d <<= 1)
    for (int j = 0; j < 8; ++j) mx[j] = fmaxf(mx[j], __shfl_xor(mx[j], d, 32));
  v8f ex;
  for (int j = 0; j < 8; ++j) ex[j] = __expf(acc[j] - mx[j]);
  v8f sm = ex;
  for (int d = 1; d < 8; d <<= 1)
    for (int j = 0; j < 8; ++j) sm[j] += __shfl_xor(sm[j], d, 32);
  float av[8], part = 0.f;
  for (int j = 0; j < 8; ++j) { av[j] = ex[j] / sm[j]; part += av[j]; }
  if (m16 < K_) {
    float* dst = attn + ((size_t)b * K_ + m16) * N_ + n0 + hi * 8;
    *(float4*)dst       = make_float4(av[0], av[1], av[2], av[3]);
    *(float4*)(dst + 4) = make_float4(av[4], av[5], av[6], av[7]);
    atomicAdd(&colsum[b * K_ + m16], part);
  }
}

// ---------------- updates = (attn/colsum) @ v ----------------
// grid: dim3(C/16, B) x 32
__global__ void __launch_bounds__(32) updates_kernel(
    const float* __restrict__ attn, const float* __restrict__ colsum,
    const bf16* __restrict__ vT, float* __restrict__ upd) {
  const int l = threadIdx.x & 31, m16 = l & 15, hi = l >> 4;
  const int ct = blockIdx.x, b = blockIdx.y;
  const float scale = (m16 < K_) ? 1.f / (colsum[b * K_ + m16] + 1e-5f) : 0.f;
  const float* arow = attn + (size_t)(b * K_ + (m16 < K_ ? m16 : 0)) * N_; // scale=0 kills pad rows
  const bf16* vcol = vT + (size_t)b * C_ * N_ + (size_t)(ct * 16 + m16) * N_;
  v8f acc = vzero8();
  for (int ntb = 0; ntb < N_ / 32; ++ntb) {
    int nb = ntb * 32;
    acc = wmma_bf16(load_a_f32(arow + nb, hi, scale),
                    load16bf(vcol + nb + hi * 16), acc);
  }
  if (hi == 0)
    for (int j = 0; j < 8; ++j)
      upd[(size_t)(b * K_ + j) * C_ + ct * 16 + m16] = acc[j];
}

// ---------------- generic GEMM: D = act(A_f32 @ Bw^T + bias) (+resid) ----------------
// grid: dim3(Nout/16, M/16) x 32
__global__ void __launch_bounds__(32) gemm_kernel(
    const float* __restrict__ A, const bf16* __restrict__ Bw,
    const float* __restrict__ bias, const float* __restrict__ resid,
    float* __restrict__ D, int Nout, int Kc, int act) {
  const int l = threadIdx.x & 31, m16 = l & 15, hi = l >> 4;
  const int nt = blockIdx.x, mt = blockIdx.y;
  const float* arow = A + (size_t)(mt * 16 + m16) * Kc;
  const bf16* brow = Bw + (size_t)(nt * 16 + m16) * Kc;
  v8f acc = vzero8();
  for (int kt = 0; kt < Kc / 32; ++kt)
    acc = wmma_bf16(load_a_f32(arow + kt * 32, hi, 1.f),
                    load16bf(brow + kt * 32 + hi * 16), acc);
  const int c = nt * 16 + m16;
  const float bi = bias ? bias[c] : 0.f;
  for (int j = 0; j < 8; ++j) {
    int row = mt * 16 + hi * 8 + j;
    float v = acc[j] + bi;
    if (act) v = 0.5f * v * (1.f + erff(v * 0.70710678118654752f)); // exact GELU
    if (resid) v += resid[(size_t)row * Nout + c];
    D[(size_t)row * Nout + c] = v;
  }
}

// ---------------- GRU gate combine ----------------
__global__ void __launch_bounds__(256) gru_kernel(
    const float* __restrict__ gi, const float* __restrict__ gh,
    const float* __restrict__ hprev, float* __restrict__ h) {
  int idx = blockIdx.x * 256 + threadIdx.x;        // 65536 = 256 rows x 256 cols
  int row = idx >> 8, c = idx & 255;
  size_t o = (size_t)row * 768 + c;
  float ir = gi[o], iz = gi[o + 256], in_ = gi[o + 512];
  float hr = gh[o], hz = gh[o + 256], hn  = gh[o + 512];
  float r = 1.f / (1.f + __expf(-(ir + hr)));
  float z = 1.f / (1.f + __expf(-(iz + hz)));
  float n = tanhf(in_ + r * hn);
  h[idx] = (1.f - z) * n + z * hprev[idx];
}

// ---------------- row-wise LayerNorm (M rows x 256) ----------------
__global__ void __launch_bounds__(256) lnrows_kernel(
    const float* __restrict__ X, const float* __restrict__ g,
    const float* __restrict__ bb, float* __restrict__ Y) {
  int wave = threadIdx.x >> 5, l = threadIdx.x & 31;
  int row = blockIdx.x * 8 + wave;
  const float* xr = X + (size_t)row * C_ + l * 8;
  float4 p0 = *(const float4*)xr, p1 = *(const float4*)(xr + 4);
  float v[8] = {p0.x,p0.y,p0.z,p0.w,p1.x,p1.y,p1.z,p1.w};
  float s = 0.f, s2 = 0.f;
  for (int i = 0; i < 8; ++i) { s += v[i]; s2 += v[i]*v[i]; }
  wave_red2(s, s2);
  float mean = s * (1.f / C_);
  float rstd = rsqrtf(s2 * (1.f / C_) - mean * mean + 1e-5f);
  for (int i = 0; i < 8; ++i) {
    int c = l * 8 + i;
    Y[(size_t)row * C_ + c] = (v[i] - mean) * rstd * g[c] + bb[c];
  }
}

extern "C" void kernel_launch(void* const* d_in, const int* in_sizes, int n_in,
                              void* d_out, int out_size, void* d_ws, size_t ws_size,
                              hipStream_t stream) {
  const float* x      = (const float*)d_in[0];
  const float* slots  = (const float*)d_in[1];
  const float* ln_in_g= (const float*)d_in[2];
  const float* ln_in_b= (const float*)d_in[3];
  const float* Wq     = (const float*)d_in[4];
  const float* Wk     = (const float*)d_in[5];
  const float* Wv     = (const float*)d_in[6];
  const float* ln0_g  = (const float*)d_in[7];
  const float* ln0_b  = (const float*)d_in[8];
  const float* Wih    = (const float*)d_in[9];
  const float* Whh    = (const float*)d_in[10];
  const float* bih    = (const float*)d_in[11];
  const float* bhh    = (const float*)d_in[12];
  const float* ln1_g  = (const float*)d_in[13];
  const float* ln1_b  = (const float*)d_in[14];
  const float* W1     = (const float*)d_in[15];
  const float* b1     = (const float*)d_in[16];
  const float* W2     = (const float*)d_in[17];
  const float* b2     = (const float*)d_in[18];

  float* out_slots = (float*)d_out;
  float* out_attn  = (float*)d_out + (size_t)B_ * K_ * C_;

  char* cur = (char*)d_ws;
  auto carve = [&](size_t bytes) -> char* {
    char* p = cur; cur += (bytes + 255) & ~(size_t)255; return p;
  };
  bf16* kbf    = (bf16*)carve((size_t)B_*N_*C_*2);
  bf16* vT     = (bf16*)carve((size_t)B_*C_*N_*2);
  bf16* qbf    = (bf16*)carve((size_t)B_*16*C_*2);
  bf16* Wq_bf  = (bf16*)carve((size_t)C_*C_*2);
  bf16* Wk_bf  = (bf16*)carve((size_t)C_*C_*2);
  bf16* Wv_bf  = (bf16*)carve((size_t)C_*C_*2);
  bf16* Wih_bf = (bf16*)carve((size_t)3*C_*C_*2);
  bf16* Whh_bf = (bf16*)carve((size_t)3*C_*C_*2);
  bf16* W1_bf  = (bf16*)carve((size_t)F_*C_*2);
  bf16* W2_bf  = (bf16*)carve((size_t)C_*F_*2);
  float* colsum  = (float*)carve((size_t)B_*K_*4);
  float* upd     = (float*)carve((size_t)B_*K_*C_*4);
  float* gi      = (float*)carve((size_t)B_*K_*3*C_*4);
  float* gh      = (float*)carve((size_t)B_*K_*3*C_*4);
  float* hbuf    = (float*)carve((size_t)B_*K_*C_*4);
  float* lnbuf   = (float*)carve((size_t)B_*K_*C_*4);
  float* ffbuf   = (float*)carve((size_t)B_*K_*F_*4);
  float* slotsbuf= (float*)carve((size_t)B_*K_*C_*4);

  auto cvt = [&](const float* s, bf16* d, int n) {
    cvt_kernel<<<(n + 255) / 256, 256, 0, stream>>>(s, d, n);
  };
  cvt(Wq,  Wq_bf,  C_*C_);
  cvt(Wk,  Wk_bf,  C_*C_);
  cvt(Wv,  Wv_bf,  C_*C_);
  cvt(Wih, Wih_bf, 3*C_*C_);
  cvt(Whh, Whh_bf, 3*C_*C_);
  cvt(W1,  W1_bf,  F_*C_);
  cvt(W2,  W2_bf,  C_*F_);

  kv_kernel<<<(B_*N_)/64, 128, 0, stream>>>(x, ln_in_g, ln_in_b, Wk_bf, Wv_bf, kbf, vT);

  const float* prev = slots;
  for (int t = 0; t < 3; ++t) {
    q_kernel<<<dim3(C_/16, B_), 32, 0, stream>>>(prev, ln0_g, ln0_b, Wq_bf, qbf, colsum);
    logits_kernel<<<dim3(N_/16, B_), 32, 0, stream>>>(kbf, qbf, out_attn, colsum);
    updates_kernel<<<dim3(C_/16, B_), 32, 0, stream>>>(out_attn, colsum, vT, upd);
    gemm_kernel<<<dim3(3*C_/16, 16), 32, 0, stream>>>(upd,  Wih_bf, bih, nullptr, gi, 3*C_, C_, 0);
    gemm_kernel<<<dim3(3*C_/16, 16), 32, 0, stream>>>(prev, Whh_bf, bhh, nullptr, gh, 3*C_, C_, 0);
    gru_kernel<<<(B_*K_*C_)/256, 256, 0, stream>>>(gi, gh, prev, hbuf);
    lnrows_kernel<<<(B_*K_)/8, 256, 0, stream>>>(hbuf, ln1_g, ln1_b, lnbuf);
    gemm_kernel<<<dim3(F_/16, 16), 32, 0, stream>>>(lnbuf, W1_bf, b1, nullptr, ffbuf, F_, C_, 1);
    float* dst = (t == 2) ? out_slots : slotsbuf;
    gemm_kernel<<<dim3(C_/16, 16), 32, 0, stream>>>(ffbuf, W2_bf, b2, hbuf, dst, C_, F_, 0);
    prev = dst;
  }
}
